// MotifConv_10153302687996
// MI455X (gfx1250) — matrix operand order repaired and checked by reading, not compiled, not run on hardware
//
#include <hip/hip_runtime.h>

// ---------------------------------------------------------------------------
// MotifConv for MI455X (gfx1250, wave32).
//  - scatters: global_atomic_add_f32 (L2-resident gather source)
//  - base GEMMs: V_WMMA_F32_16X16X4_F32 (full f32 precision)
//  - 13 motif GEMMs + attention: V_WMMA_F32_16X16X32_BF16, A staged in LDS
// ---------------------------------------------------------------------------

#define NN     100000
#define EE     1600000
#define MNNZ   800000
#define NMOTIF 13

typedef __attribute__((ext_vector_type(2)))  float  v2f;
typedef __attribute__((ext_vector_type(4)))  float  v4f;
typedef __attribute__((ext_vector_type(8)))  float  v8f;
typedef __attribute__((ext_vector_type(4)))  __bf16 v4bf;
typedef __attribute__((ext_vector_type(8)))  __bf16 v8bf;
typedef __attribute__((ext_vector_type(16))) __bf16 v16bf;

// -------- weight prep: motif_W [13][832][64] f32 -> Wt bf16 [13][64][832],
//          wa [64][64] f32 -> wat bf16 [64][64] (both "column-major" for B) --
__global__ void prep_weights(const float* __restrict__ mW,
                             const float* __restrict__ wa,
                             __bf16* __restrict__ Wt,
                             __bf16* __restrict__ wat) {
  const int idx = blockIdx.x * 256 + threadIdx.x;
  const int NW = NMOTIF * 832 * 64;        // 692224
  if (idx < NW) {
    const int i   = idx / (832 * 64);
    const int r   = idx % (832 * 64);
    const int k   = r >> 6;
    const int col = r & 63;
    Wt[(size_t)i * (64 * 832) + (size_t)col * 832 + k] = (__bf16)mW[idx];
  } else if (idx < NW + 64 * 64) {
    const int r   = idx - NW;              // r = k*64 + col
    const int k   = r >> 6;
    const int col = r & 63;
    wat[col * 64 + k] = (__bf16)wa[r];
  }
}

// -------- xw = x@W_rel, xr = x@root via f32 WMMA (K=4 steps) ---------------
__global__ void __launch_bounds__(64)
lin_wmma(const float* __restrict__ x, const float* __restrict__ Wrel,
         const float* __restrict__ root,
         float* __restrict__ xw, float* __restrict__ xr) {
  const int lane  = threadIdx.x & 31;
  const int wv    = threadIdx.x >> 5;
  const int base  = (blockIdx.x * 2 + wv) * 16;   // 6250 tiles exactly cover N
  const int mrow  = lane & 15;
  const int khalf = (lane >> 4) << 1;             // 0 or 2

  const v8f zero = {0.f, 0.f, 0.f, 0.f, 0.f, 0.f, 0.f, 0.f};
  v8f accW[4], accR[4];
#pragma unroll
  for (int t = 0; t < 4; ++t) { accW[t] = zero; accR[t] = zero; }

#pragma unroll
  for (int kk = 0; kk < 64; kk += 4) {
    const int ka = kk + khalf;
    v2f a = *(const v2f*)(x + (size_t)(base + mrow) * 64 + ka);
#pragma unroll
    for (int t = 0; t < 4; ++t) {
      const int col = t * 16 + mrow;
      v2f bW = {Wrel[ka * 64 + col], Wrel[(ka + 1) * 64 + col]};
      v2f bR = {root[ka * 64 + col], root[(ka + 1) * 64 + col]};
      accW[t] = __builtin_amdgcn_wmma_f32_16x16x4_f32(false, a, false, bW, 0,
                                                      accW[t], false, false);
      accR[t] = __builtin_amdgcn_wmma_f32_16x16x4_f32(false, a, false, bR, 0,
                                                      accR[t], false, false);
    }
  }
  const int rbase = base + ((lane >> 4) << 3);
#pragma unroll
  for (int t = 0; t < 4; ++t) {
    const int col = t * 16 + mrow;
#pragma unroll
    for (int v = 0; v < 8; ++v) {
      xw[(size_t)(rbase + v) * 64 + col] = accW[t][v];
      xr[(size_t)(rbase + v) * 64 + col] = accR[t][v];
    }
  }
}

// -------- edge scatter: hs[dst] += w_e * xw[src]; deg[dst] += 1 ------------
__global__ void edge_scatter(const float* __restrict__ xw,
                             const int* __restrict__ ei,
                             const float* __restrict__ ew,
                             float* __restrict__ hs, float* __restrict__ deg) {
  const int tid = blockIdx.x * 256 + threadIdx.x;  // EE*16 threads exactly
  const int e = tid >> 4;
  const int c = (tid & 15) << 2;
  const int s = ei[e];
  const int d = ei[EE + e];
  const float w = ew[e];
  v4f v = *(const v4f*)(xw + (size_t)s * 64 + c);
  float* o = hs + (size_t)d * 64 + c;
  atomicAdd(o + 0, v[0] * w);
  atomicAdd(o + 1, v[1] * w);
  atomicAdd(o + 2, v[2] * w);
  atomicAdd(o + 3, v[3] * w);
  if ((tid & 15) == 0) atomicAdd(deg + d, 1.0f);
}

// -------- h = hs * (1/deg) + xr + bias -------------------------------------
__global__ void finalize_h(const float* __restrict__ hs,
                           const float* __restrict__ xr,
                           const float* __restrict__ deg,
                           const float* __restrict__ bias,
                           float* __restrict__ hout) {
  const int idx = blockIdx.x * 256 + threadIdx.x;  // N*64 exactly
  const int n = idx >> 6;
  const int c = idx & 63;
  const float dg = deg[n];
  const float nm = dg > 0.0f ? 1.0f / dg : 0.0f;
  hout[idx] = hs[idx] * nm + xr[idx] + bias[c];
}

// -------- 13 motif scatters, accumulated directly into d_out ---------------
__global__ void motif_scatter(const float* __restrict__ h,
                              const int* __restrict__ msrc,
                              const int* __restrict__ mdst,
                              const float* __restrict__ mval,
                              float* __restrict__ out) {
  const int tid = blockIdx.x * 256 + threadIdx.x;  // 13*MNNZ*16 threads exactly
  const int g = tid >> 4;
  const int i = g / MNNZ;
  const int m = g % MNNZ;
  const int c = (tid & 15) << 2;
  const size_t off = (size_t)i * MNNZ + m;
  const int s = msrc[off];
  const int d = mdst[off];
  const float w = mval[off];
  v4f v = *(const v4f*)(h + (size_t)s * 64 + c);
  float* o = out + (size_t)d * 832 + i * 64 + c;
  atomicAdd(o + 0, v[0] * w);
  atomicAdd(o + 1, v[1] * w);
  atomicAdd(o + 2, v[2] * w);
  atomicAdd(o + 3, v[3] * w);
}

// -------- fused: per 16-node tile, stage [16 x 896] bf16 in LDS, then for
//          each motif: mw = rst_i@wa, c = compress@motif_W[i-1], attention,
//          out = att*(mw-c). Overwrites d_out in place (row-local dep only). -
__global__ void __launch_bounds__(64)
motif_attn(const float* __restrict__ h, float* __restrict__ out,
           const __bf16* __restrict__ Wt, const __bf16* __restrict__ wat,
           const float* __restrict__ ba, const float* __restrict__ mb) {
  __shared__ __bf16 lds[2 * 16 * 904];             // 904 stride: 2-way max conflicts
  const int lane = threadIdx.x & 31;
  const int wv   = threadIdx.x >> 5;
  const int base = (blockIdx.x * 2 + wv) * 16;     // 6250 tiles == N/16 exactly
  __bf16* L = lds + wv * (16 * 904);

  // stage 16 rows x 896 ch: block0 = h, blocks 1..13 = aggregations (in out)
  for (int j = lane * 4; j < 16 * 896; j += 32 * 4) {
    const int row = j / 896;
    const int c   = j % 896;
    v4f val;
    if (c < 64) val = *(const v4f*)(h   + (size_t)(base + row) * 64  + c);
    else        val = *(const v4f*)(out + (size_t)(base + row) * 832 + (c - 64));
    v4bf b;
    b[0] = (__bf16)val[0]; b[1] = (__bf16)val[1];
    b[2] = (__bf16)val[2]; b[3] = (__bf16)val[3];
    *(v4bf*)(L + row * 904 + c) = b;
  }
  asm volatile("s_wait_dscnt 0" ::: "memory");

  const int mrow  = lane & 15;
  const int koffA = (lane >> 4) << 4;   // A operand: 16 consecutive K per lane
  const int koffB = (lane >> 4) << 4;   // B operand: 16 consecutive K per lane
  typedef union { v16bf v; v8bf h2[2]; } abu;
  const v8f zero = {0.f, 0.f, 0.f, 0.f, 0.f, 0.f, 0.f, 0.f};

  for (int i = 1; i <= NMOTIF; ++i) {
    v8f acc_c[4], acc_m[4];
#pragma unroll
    for (int t = 0; t < 4; ++t) { acc_c[t] = zero; acc_m[t] = zero; }

    // mw = rst_i @ wa   (K = 64 -> two 16x16x32 steps per column tile)
#pragma unroll
    for (int kf = 0; kf < 2; ++kf) {
      abu a;
      const __bf16* ap = L + mrow * 904 + i * 64 + kf * 32 + koffA;
      a.h2[0] = *(const v8bf*)(ap);
      a.h2[1] = *(const v8bf*)(ap + 8);
#pragma unroll
      for (int t = 0; t < 4; ++t) {
        const int col = t * 16 + mrow;
        v16bf b = *(const v16bf*)(wat + col * 64 + kf * 32 + koffB);
        acc_m[t] = __builtin_amdgcn_wmma_f32_16x16x32_bf16(false, a.v, false, b,
                                                           0, acc_m[t], false, false);
      }
    }

    // c = compress @ motif_W[i-1]   (13 K-blocks, skipping block i)
    const __bf16* Wti = Wt + (size_t)(i - 1) * (64 * 832);
    for (int jj = 0; jj < 13; ++jj) {
      const int blk = (jj < i) ? jj : jj + 1;
#pragma unroll
      for (int kf = 0; kf < 2; ++kf) {
        abu a;
        const __bf16* ap = L + mrow * 904 + blk * 64 + kf * 32 + koffA;
        a.h2[0] = *(const v8bf*)(ap);
        a.h2[1] = *(const v8bf*)(ap + 8);
#pragma unroll
        for (int t = 0; t < 4; ++t) {
          const int col = t * 16 + mrow;
          v16bf b = *(const v16bf*)(Wti + (size_t)col * 832 + jj * 64 +
                                    kf * 32 + koffB);
          acc_c[t] = __builtin_amdgcn_wmma_f32_16x16x32_bf16(false, a.v, false, b,
                                                             0, acc_c[t], false, false);
        }
      }
    }

    // attention: sigmoid(row-dot(mw, c)) via 16-lane xor reduction
    float s[8];
#pragma unroll
    for (int v = 0; v < 8; ++v) s[v] = 0.0f;
#pragma unroll
    for (int t = 0; t < 4; ++t) {
      const int col = t * 16 + mrow;
      const float bav = ba[col];
      const float mbv = mb[(i - 1) * 64 + col];
#pragma unroll
      for (int v = 0; v < 8; ++v)
        s[v] += (acc_m[t][v] + bav) * (acc_c[t][v] + mbv);
    }
#pragma unroll
    for (int v = 0; v < 8; ++v) {
      float t0 = s[v];
      t0 += __shfl_xor(t0, 1, 16);
      t0 += __shfl_xor(t0, 2, 16);
      t0 += __shfl_xor(t0, 4, 16);
      t0 += __shfl_xor(t0, 8, 16);
      s[v] = 1.0f / (1.0f + __expf(-t0));
    }

    const int rbase = base + ((lane >> 4) << 3);
#pragma unroll
    for (int t = 0; t < 4; ++t) {
      const int col = t * 16 + mrow;
      const float bav = ba[col];
      const float mbv = mb[(i - 1) * 64 + col];
#pragma unroll
      for (int v = 0; v < 8; ++v) {
        const float mwv = acc_m[t][v] + bav;
        const float cv  = acc_c[t][v] + mbv;
        out[(size_t)(rbase + v) * 832 + (i - 1) * 64 + col] = s[v] * (mwv - cv);
      }
    }
  }
}

// ---------------------------------------------------------------------------
extern "C" void kernel_launch(void* const* d_in, const int* in_sizes, int n_in,
                              void* d_out, int out_size, void* d_ws, size_t ws_size,
                              hipStream_t stream) {
  (void)in_sizes; (void)n_in; (void)out_size; (void)ws_size;
  const float* x    = (const float*)d_in[0];
  const int*   ei   = (const int*)  d_in[1];
  const float* ew   = (const float*)d_in[2];
  const int*   msrc = (const int*)  d_in[3];
  const int*   mdst = (const int*)  d_in[4];
  const float* mval = (const float*)d_in[5];
  const float* Wrel = (const float*)d_in[6];
  const float* root = (const float*)d_in[7];
  const float* bias = (const float*)d_in[8];
  const float* wa   = (const float*)d_in[9];
  const float* ba   = (const float*)d_in[10];
  const float* mW   = (const float*)d_in[11];
  const float* mb   = (const float*)d_in[12];
  float* out = (float*)d_out;

  float* ws  = (float*)d_ws;
  float* xw  = ws;                          // N*64 f32; later reused as h
  float* xr  = xw + (size_t)NN * 64;        // N*64 f32
  float* hs  = xr + (size_t)NN * 64;        // N*64 f32 (edge accumulator)
  float* deg = hs + (size_t)NN * 64;        // N f32
  __bf16* Wt  = (__bf16*)(deg + NN);        // 13*64*832 bf16
  __bf16* wat = Wt + (size_t)NMOTIF * 64 * 832;  // 64*64 bf16

  hipMemsetAsync(hs,  0, (size_t)NN * 64 * sizeof(float), stream);
  hipMemsetAsync(deg, 0, (size_t)NN * sizeof(float), stream);
  hipMemsetAsync(out, 0, (size_t)NN * 832 * sizeof(float), stream);

  prep_weights <<<2720,   256, 0, stream>>>(mW, wa, Wt, wat);
  lin_wmma     <<<3125,    64, 0, stream>>>(x, Wrel, root, xw, xr);
  edge_scatter <<<100000, 256, 0, stream>>>(xw, ei, ew, hs, deg);
  finalize_h   <<<25000,  256, 0, stream>>>(hs, xr, deg, bias, xw);  // h := xw buf
  motif_scatter<<<650000, 256, 0, stream>>>(xw, msrc, mdst, mval, out);
  motif_attn   <<<3125,    64, 0, stream>>>(xw, out, Wt, wat, ba, mb);
}
